// TransformerBlock_25872882991987
// MI455X (gfx1250) — matrix operand order repaired and checked by reading, not compiled, not run on hardware
//
#include <hip/hip_runtime.h>
#include <hip/hip_bf16.h>
#include <math.h>

// ---------------------------------------------------------------------------
// Transformer block (LN -> linear attention -> LN -> FFN) for MI455X gfx1250.
// GEMMs run as Out^T = W * X^T on v_wmma_f32_16x16x32_f16.
//  - LN affine folded into pre-converted f16 weights.
//  - All GEMM B operands stored f16 *transposed* [n][C] (channel-contiguous):
//    a B fragment is exactly two global_load_b128 per lane, zero packing ops.
//  - w1 epilogue / mid / hidden tensors written as vectorized f16 [n][C].
// ---------------------------------------------------------------------------

typedef __attribute__((ext_vector_type(16))) _Float16 v16h;
typedef __attribute__((ext_vector_type(8)))  _Float16 v8h;
typedef __attribute__((ext_vector_type(8)))  float    v8f;

#define HW      65536      // 256*256
#define HWSH    16
#define HWMASK  65535
#define NBATCH  4
#define CCH     192
#define NHEADS  8
#define HD      24
#define NTOT    (NBATCH * HW)   // 262144 pixels

static __device__ __forceinline__ size_t chw_off(int Cdim, int c, int n) {
  return ((size_t)((n >> HWSH) * Cdim + c)) * (size_t)HW + (size_t)(n & HWMASK);
}

static __device__ __forceinline__ float gelu_f(float x) {
  return 0.5f * x * (1.0f + erff(x * 0.7071067811865475f));
}

static __device__ __forceinline__ v8f wmma_f16(v16h a, v16h b, v8f c) {
  return __builtin_amdgcn_wmma_f32_16x16x32_f16(false, a, false, b, (short)0, c,
                                                false, false);
}

// ---------------------------------------------------------------------------
// Weight prep: Wh[cout][k] = f16( W[cout][k] * g[k] ), bias2 = bias + W @ b.
// (g/b null -> plain f32->f16 conversion, bias passthrough.)
// ---------------------------------------------------------------------------
__global__ __launch_bounds__(256) void foldw_k(const float* __restrict__ W,
                                               const float* __restrict__ bias,
                                               const float* __restrict__ g,
                                               const float* __restrict__ b,
                                               _Float16* __restrict__ Wh,
                                               float* __restrict__ bias2,
                                               int Cin, int Cout) {
  int co = blockIdx.x * 256 + threadIdx.x;
  if (co >= Cout) return;
  const float* wr = W + (size_t)co * Cin;
  _Float16* wo = Wh + (size_t)co * Cin;
  float acc = bias[co];
  for (int k = 0; k < Cin; ++k) {
    float wv = wr[k];
    if (g) { acc += wv * b[k]; wv *= g[k]; }
    wo[k] = (_Float16)wv;
  }
  bias2[co] = acc;
}

// ---------------------------------------------------------------------------
// Fused LayerNorm: per-pixel stats over channels (BCHW f32 source), then
// normalized values written as f16 transposed [n][C] in v8h chunks.
// ---------------------------------------------------------------------------
__global__ __launch_bounds__(256) void ln_f16_t_k(const float* __restrict__ src,
                                                  _Float16* __restrict__ dstT) {
  int n = blockIdx.x * 256 + threadIdx.x;
  if (n >= NTOT) return;
  const float* p = src + (size_t)(n >> HWSH) * CCH * (size_t)HW + (n & HWMASK);
  float s = 0.f, s2 = 0.f;
  for (int c = 0; c < CCH; ++c) {
    float v = p[(size_t)c * HW];
    s += v; s2 += v * v;
  }
  float m = s * (1.0f / CCH);
  float var = s2 * (1.0f / CCH) - m * m;
  float r = rsqrtf(var + 1e-5f);
  float nmr = -m * r;
  _Float16* d = dstT + (size_t)n * CCH;
  for (int c0 = 0; c0 < CCH; c0 += 8) {
    v8h pk;
#pragma unroll
    for (int j = 0; j < 8; ++j)
      pk[j] = (_Float16)fmaf(p[(size_t)(c0 + j) * HW], r, nmr);
    *(v8h*)(d + c0) = pk;
  }
}

// ---------------------------------------------------------------------------
// WMMA GEMM engine: out^T = Wh[Cout,Cin](f16) * srcT[n][Cin](f16)
// (+bias)(+gelu)(+residual).  Wave tile: 32 Cout x 64 pixels, 8 wmma/K-step.
// B fragment = two b128 loads (channels kt+hi8..+7 and kt+16+hi8..+7 at the lane's pixel).
// grid.x = Cout/32, grid.y = NTOT/512 (8 waves/block).
// F16T: store f16 transposed [n][Cout] (one b128 per 8 outputs).
// ---------------------------------------------------------------------------
template <int CIN, bool RES, bool GELU, bool F16T>
__global__ __launch_bounds__(256)
void gemm_chw_k(const _Float16* __restrict__ Wh, const float* __restrict__ bias,
                const _Float16* __restrict__ srcT,
                const float* __restrict__ resid,
                void* __restrict__ outv, int Cout) {
  const int lane  = threadIdx.x & 31;
  const int wave  = threadIdx.x >> 5;
  const bool hi   = lane >= 16;
  const int hi8   = hi ? 8 : 0;
  const int hi16  = hi ? 16 : 0;
  const int col   = lane & 15;
  const int cout0 = blockIdx.x * 32;
  const int n0    = (blockIdx.y * 8 + wave) * 64;

  // A rows (ISA 16-bit A layout: lanes 0-15 row M, K {0..7,16..23}; lanes
  // 16-31 same row, K {8..15,24..31}) -> two b128 loads per fragment.
  const _Float16* wrow0 = Wh + (size_t)(cout0 + col) * CIN;
  const _Float16* wrow1 = wrow0 + (size_t)16 * CIN;

  // Per pixel-subtile base pointers into the transposed f16 activations.
  const _Float16* sp[4];
#pragma unroll
  for (int t = 0; t < 4; ++t)
    sp[t] = srcT + (size_t)(t * 16 + n0 + col) * CIN;

  v8f acc0[4] = {};
  v8f acc1[4] = {};

  for (int kt = 0; kt < CIN; kt += 32) {
    v8h l0 = *(const v8h*)(wrow0 + kt + hi8);
    v8h h0 = *(const v8h*)(wrow0 + kt + 16 + hi8);
    v8h l1 = *(const v8h*)(wrow1 + kt + hi8);
    v8h h1 = *(const v8h*)(wrow1 + kt + 16 + hi8);
    v16h a0 = __builtin_shufflevector(l0, h0, 0, 1, 2, 3, 4, 5, 6, 7,
                                      8, 9, 10, 11, 12, 13, 14, 15);
    v16h a1 = __builtin_shufflevector(l1, h1, 0, 1, 2, 3, 4, 5, 6, 7,
                                      8, 9, 10, 11, 12, 13, 14, 15);
#pragma unroll
    for (int t = 0; t < 4; ++t) {
      const _Float16* bsrc = sp[t] + kt + hi8;
      v8h b_lo = *(const v8h*)bsrc;
      v8h b_hi = *(const v8h*)(bsrc + 16);
      v16h bfrag = __builtin_shufflevector(b_lo, b_hi, 0, 1, 2, 3, 4, 5, 6, 7,
                                           8, 9, 10, 11, 12, 13, 14, 15);
      acc0[t] = wmma_f16(a0, bfrag, acc0[t]);
      acc1[t] = wmma_f16(a1, bfrag, acc1[t]);
    }
  }

  // ---- Epilogue --------------------------------------------------------
#pragma unroll
  for (int t = 0; t < 4; ++t) {
    int nb = t * 16 + (n0 + col);
#pragma unroll
    for (int half = 0; half < 2; ++half) {
      int cbase = cout0 + half * 16 + hi8;
      v8f a = half ? acc1[t] : acc0[t];
      const float* bp = bias + cbase;
      if (F16T) {
        v8h pk;
#pragma unroll
        for (int r = 0; r < 8; ++r) {
          float v = a[r] + bp[r];
          if (GELU) v = gelu_f(v);
          pk[r] = (_Float16)v;
        }
        *(v8h*)((_Float16*)outv + (size_t)nb * Cout + cbase) = pk;
      } else {
        size_t obase = (size_t)(nb >> HWSH) * Cout * (size_t)HW + (nb & HWMASK);
        const float* rp = resid + obase + (size_t)cbase * HW;  // unused if !RES
#pragma unroll
        for (int r = 0; r < 8; ++r) {
          float v = a[r] + bp[r];
          if (GELU) v = gelu_f(v);
          if (RES) v += rp[(size_t)r * HW];
          ((float*)outv)[obase + (size_t)(cbase + r) * HW] = v;
        }
      }
    }
  }
}

// ---------------------------------------------------------------------------
// Softmax of k over pixels: one workgroup per (batch, k-row). Rows 192..383.
// ---------------------------------------------------------------------------
__global__ __launch_bounds__(256) void k_softmax_k(float* __restrict__ qkv) {
  int b = blockIdx.x / CCH;
  int r = blockIdx.x - b * CCH;
  float* p = qkv + ((size_t)b * 576 + 192 + r) * (size_t)HW;
  __shared__ float red[256];
  int tid = threadIdx.x;

  float m = -3.4e38f;
  for (int i = tid; i < HW; i += 256) m = fmaxf(m, p[i]);
  red[tid] = m; __syncthreads();
  for (int s = 128; s > 0; s >>= 1) {
    if (tid < s) red[tid] = fmaxf(red[tid], red[tid + s]);
    __syncthreads();
  }
  m = red[0]; __syncthreads();

  float acc = 0.f;
  for (int i = tid; i < HW; i += 256) acc += expf(p[i] - m);
  red[tid] = acc; __syncthreads();
  for (int s = 128; s > 0; s >>= 1) {
    if (tid < s) red[tid] += red[tid + s];
    __syncthreads();
  }
  float inv = 1.0f / red[0]; __syncthreads();

  for (int i = tid; i < HW; i += 256) p[i] = expf(p[i] - m) * inv;
}

// ---------------------------------------------------------------------------
// context[b,h,ch,d] = sum_p k[b,h,ch,p] * v[b,h,d,p]   (24x24 per (b,h))
// ---------------------------------------------------------------------------
__global__ void ctx_zero_k(float* __restrict__ ctx) {
  int i = blockIdx.x * 256 + threadIdx.x;
  if (i < NBATCH * NHEADS * HD * HD) ctx[i] = 0.f;
}

__global__ void context_k(const float* __restrict__ qkv, float* __restrict__ ctx) {
  int bh = blockIdx.x;
  int b = bh >> 3, h = bh & 7;
  int d = threadIdx.x, ch = threadIdx.y;
  const float* kp = qkv + ((size_t)b * 576 + 192 + h * HD + ch) * (size_t)HW;
  const float* vp = qkv + ((size_t)b * 576 + 384 + h * HD + d) * (size_t)HW;
  int p0 = blockIdx.y * (HW / 16);
  float acc = 0.f;
  for (int i = 0; i < HW / 16; ++i) acc += kp[p0 + i] * vp[p0 + i];
  atomicAdd(&ctx[(bh * HD + ch) * HD + d], acc);
}

// ---------------------------------------------------------------------------
// Fused: q-softmax over head_dim, then midT[n][h*24+d] = (ctx^T q)  (f16).
// ---------------------------------------------------------------------------
__global__ __launch_bounds__(256) void q_attn_k(const float* __restrict__ qkv,
                                                const float* __restrict__ ctx,
                                                _Float16* __restrict__ midT) {
  int idx = blockIdx.x * 256 + threadIdx.x;
  if (idx >= NHEADS * NTOT) return;
  int h = idx / NTOT;
  int n = idx - h * NTOT;
  int b = n >> HWSH;
  const float* qp = qkv + chw_off(576, h * HD, n);
  const float* c = ctx + ((b * NHEADS + h) * HD) * HD;

  float qv[HD];
  float m = -3.4e38f;
#pragma unroll
  for (int ch = 0; ch < HD; ++ch) {
    qv[ch] = qp[(size_t)ch * HW];
    m = fmaxf(m, qv[ch]);
  }
  float s = 0.f;
#pragma unroll
  for (int ch = 0; ch < HD; ++ch) { qv[ch] = expf(qv[ch] - m); s += qv[ch]; }
  float inv = 1.0f / s;
#pragma unroll
  for (int ch = 0; ch < HD; ++ch) qv[ch] *= inv;

  _Float16 tmp[HD];
#pragma unroll
  for (int d = 0; d < HD; ++d) {
    float acc = 0.f;
#pragma unroll
    for (int ch = 0; ch < HD; ++ch) acc += c[ch * HD + d] * qv[ch];
    tmp[d] = (_Float16)acc;
  }
  _Float16* mp = midT + (size_t)n * CCH + h * HD;
#pragma unroll
  for (int j = 0; j < 3; ++j) {
    v8h pk;
#pragma unroll
    for (int r = 0; r < 8; ++r) pk[r] = tmp[j * 8 + r];
    *(v8h*)(mp + j * 8) = pk;
  }
}

// ---------------------------------------------------------------------------
// Depthwise 3x3 (SAME) + GELU on the f16 transposed hidden [n][384].
// Thread = (pixel, 8-channel block): 9 b128 tap loads, 1 b128 store.
// grid = (NTOT/256, 48).
// ---------------------------------------------------------------------------
__global__ __launch_bounds__(256) void dwconv_k(const _Float16* __restrict__ hT,
                                                const float* __restrict__ w,
                                                const float* __restrict__ bias,
                                                _Float16* __restrict__ outT) {
  const int CH = 2 * CCH;  // 384
  int n = blockIdx.x * 256 + threadIdx.x;
  int c0 = blockIdx.y * 8;
  int p = n & HWMASK;
  int x0 = p & 255, y0 = p >> 8;
  size_t nbase = (size_t)(n & ~HWMASK);

  float acc[8];
#pragma unroll
  for (int j = 0; j < 8; ++j) acc[j] = bias[c0 + j];

#pragma unroll
  for (int dy = -1; dy <= 1; ++dy) {
    int yy = y0 + dy;
    if (yy < 0 || yy > 255) continue;
#pragma unroll
    for (int dx = -1; dx <= 1; ++dx) {
      int xx = x0 + dx;
      if (xx < 0 || xx > 255) continue;
      const _Float16* tp = hT + (nbase + (yy << 8) + xx) * CH + c0;
      v8h hv = *(const v8h*)tp;
#pragma unroll
      for (int j = 0; j < 8; ++j)
        acc[j] += (float)hv[j] * w[(c0 + j) * 9 + (dy + 1) * 3 + (dx + 1)];
    }
  }
  v8h pk;
#pragma unroll
  for (int j = 0; j < 8; ++j) pk[j] = (_Float16)gelu_f(acc[j]);
  *(v8h*)(outT + (size_t)n * CH + c0) = pk;
}

// ---------------------------------------------------------------------------
// Host-side orchestration.
// ---------------------------------------------------------------------------
extern "C" void kernel_launch(void* const* d_in, const int* in_sizes, int n_in,
                              void* d_out, int out_size, void* d_ws, size_t ws_size,
                              hipStream_t stream) {
  (void)in_sizes; (void)n_in; (void)out_size; (void)ws_size;

  const float* x     = (const float*)d_in[0];
  const float* ln1_g = (const float*)d_in[1];
  const float* ln1_b = (const float*)d_in[2];
  const float* qkv_w = (const float*)d_in[3];
  const float* qkv_b = (const float*)d_in[4];
  const float* out_w = (const float*)d_in[5];
  const float* out_b = (const float*)d_in[6];
  const float* ln2_g = (const float*)d_in[7];
  const float* ln2_b = (const float*)d_in[8];
  const float* w1    = (const float*)d_in[9];
  const float* b1    = (const float*)d_in[10];
  const float* dw_w  = (const float*)d_in[11];
  const float* dw_b  = (const float*)d_in[12];
  const float* w2    = (const float*)d_in[13];
  const float* b2    = (const float*)d_in[14];
  float* yout = (float*)d_out;

  char* ws = (char*)d_ws;
  const size_t SZ_QKV  = (size_t)NBATCH * 576 * HW * 4;      // 604 MB (f32)
  const size_t SZ_CN_H = (size_t)NBATCH * CCH * HW * 2;      // 100 MB (f16)
  const size_t SZ_CN   = (size_t)NBATCH * CCH * HW * 4;      // 201 MB (f32)
  const size_t SZ_HN_H = (size_t)NBATCH * 2 * CCH * HW * 2;  // 201 MB (f16)

  float*    qkv  = (float*)(ws + 0);
  size_t off = SZ_QKV;
  _Float16* midT = (_Float16*)(ws + off); off += SZ_CN_H;
  float*    y    = (float*)(ws + off);    off += SZ_CN;
  _Float16* xnT  = (_Float16*)(ws + off); off += SZ_CN_H;
  _Float16* ynT  = (_Float16*)(ws + off); off += SZ_CN_H;
  float* ctx = (float*)(ws + off); off += (size_t)NBATCH * NHEADS * HD * HD * 4;
  _Float16* qkvw_h = (_Float16*)(ws + off); off += (size_t)576 * CCH * 2;
  _Float16* outw_h = (_Float16*)(ws + off); off += (size_t)CCH * CCH * 2;
  _Float16* w1_h   = (_Float16*)(ws + off); off += (size_t)2 * CCH * CCH * 2;
  _Float16* w2_h   = (_Float16*)(ws + off); off += (size_t)CCH * 2 * CCH * 2;
  float* qkvb2 = (float*)(ws + off); off += 576 * 4;
  float* outb2 = (float*)(ws + off); off += CCH * 4;
  float* b1_2  = (float*)(ws + off); off += 2 * CCH * 4;
  float* b2_2  = (float*)(ws + off); off += CCH * 4;
  // After attention, qkv (604 MB) is dead; reuse it for the f16 FFN tensors:
  _Float16* hT  = (_Float16*)(ws + 0);        // [0, 201 MB)
  _Float16* h2T = (_Float16*)(ws + SZ_HN_H);  // [201 MB, 403 MB)

  dim3 blk256(256);
  dim3 grid_ln(NTOT / 256);
  dim3 grid_hn((NHEADS * NTOT + 255) / 256);

  // 0) weight prep: fold LN affine, convert to f16
  hipLaunchKernelGGL(foldw_k, dim3(3), blk256, 0, stream,
                     qkv_w, qkv_b, ln1_g, ln1_b, qkvw_h, qkvb2, CCH, 576);
  hipLaunchKernelGGL(foldw_k, dim3(1), blk256, 0, stream,
                     out_w, out_b, nullptr, nullptr, outw_h, outb2, CCH, CCH);
  hipLaunchKernelGGL(foldw_k, dim3(2), blk256, 0, stream,
                     w1, b1, ln2_g, ln2_b, w1_h, b1_2, CCH, 2 * CCH);
  hipLaunchKernelGGL(foldw_k, dim3(1), blk256, 0, stream,
                     w2, b2, nullptr, nullptr, w2_h, b2_2, 2 * CCH, CCH);

  // 1) LN1 (fused stats + normalize + f16 transpose)
  hipLaunchKernelGGL(ln_f16_t_k, grid_ln, blk256, 0, stream, x, xnT);
  // 2) qkv = (qkv_w.g) * xn + (qkv_b + qkv_w.b)   (WMMA)
  hipLaunchKernelGGL((gemm_chw_k<CCH, false, false, false>),
                     dim3(576 / 32, NTOT / 512), blk256, 0, stream,
                     qkvw_h, qkvb2, xnT, nullptr, qkv, 576);
  // 3) softmax(k) over pixels
  hipLaunchKernelGGL(k_softmax_k, dim3(NBATCH * CCH), blk256, 0, stream, qkv);
  // 4) context = k v^T (24x24 per batch-head)
  hipLaunchKernelGGL(ctx_zero_k, dim3((NBATCH * NHEADS * HD * HD + 255) / 256),
                     blk256, 0, stream, ctx);
  hipLaunchKernelGGL(context_k, dim3(NBATCH * NHEADS, 16), dim3(HD, HD), 0, stream,
                     qkv, ctx);
  // 5) midT = ctx^T softmax(q) -> f16 [n][C]   (fused q-softmax)
  hipLaunchKernelGGL(q_attn_k, grid_hn, blk256, 0, stream, qkv, ctx, midT);
  // 6) y = x + out_w * mid + out_b         (WMMA, fused residual)
  hipLaunchKernelGGL((gemm_chw_k<CCH, true, false, false>),
                     dim3(CCH / 32, NTOT / 512), blk256, 0, stream,
                     outw_h, outb2, midT, x, y, CCH);
  // 7) LN2 (fused stats + normalize + f16 transpose)
  hipLaunchKernelGGL(ln_f16_t_k, grid_ln, blk256, 0, stream, y, ynT);
  // 8) hT = gelu((w1.g) * yn + (b1 + w1.b)) -> f16 [n][2C]  (WMMA + GELU)
  hipLaunchKernelGGL((gemm_chw_k<CCH, false, true, true>),
                     dim3(2 * CCH / 32, NTOT / 512), blk256, 0, stream,
                     w1_h, b1_2, ynT, nullptr, hT, 2 * CCH);
  // 9) h2T = gelu(dwconv3x3(hT) + dw_b) -> f16 [n][2C]
  hipLaunchKernelGGL(dwconv_k, dim3(NTOT / 256, 48), blk256, 0, stream,
                     hT, dw_w, dw_b, h2T);
  // 10) out = y + w2 * h2 + b2             (WMMA, fused residual)
  hipLaunchKernelGGL((gemm_chw_k<2 * CCH, true, false, false>),
                     dim3(CCH / 32, NTOT / 512), blk256, 0, stream,
                     w2_h, b2_2, h2T, y, yout, CCH);
}
